// LocalAttention_backup_70480413327460
// MI455X (gfx1250) — compile-verified, matrix operand order checked by reading, not run on hardware
//
#include <hip/hip_runtime.h>
#include <hip/hip_bf16.h>
#include <cstdint>

#define GS     64            // grid side S
#define NTOK   4096          // S*S tokens per batch image
#define BATCH  8
#define M_TOT  (BATCH*NTOK)  // 32768 tokens per step
#define DD     48
#define EE     50
#define EP     64            // padded feature dim
#define T_OBS  10
#define T_TOT  59

typedef __attribute__((ext_vector_type(16))) __bf16 v16bf;
typedef __attribute__((ext_vector_type(8)))  float  v8f;

union FragA { uint32_t u[8]; v16bf v; };
union FragC { float    f[8]; v8f   v; };

__device__ __forceinline__ uint16_t f2bf(float f) {
  uint32_t u = __float_as_uint(f);
  uint32_t r = (u + 0x7fffu + ((u >> 16) & 1u)) >> 16;   // round-to-nearest-even
  return (uint16_t)r;
}
__device__ __forceinline__ uint32_t pack2bf(float lo, float hi) {
  return (uint32_t)f2bf(lo) | ((uint32_t)f2bf(hi) << 16);
}
__device__ __forceinline__ int clampi(int v, int lo, int hi) {
  return v < lo ? lo : (v > hi ? hi : v);
}

// ---------------------------------------------------------------------------
// Weight pre-pack: B-fragment layout (wave32, 16x16x32 bf16).
// Lane (hv=lane>>4, ls=lane&15): column N = nt*16+ls; VGPR i holds bf16 pair
// K = base, base+1 with base = (i<4?0:16) + hv*8 + (i&3)*2  (+ kc*32).
// frag id f = mat*8 + nt*2 + kc ; storage: packw[(f*32 + lane)*8 + i]
// mats: 0 Wq, 1 Wk, 2 Wv, 3 Wout, 4 Wsa, 5 W2. Out-of-range -> 0 (padding).
// ---------------------------------------------------------------------------
__global__ void prepack(const float* __restrict__ inw,  const float* __restrict__ outw,
                        const float* __restrict__ saw,  const float* __restrict__ w2,
                        const float* __restrict__ inb,  const float* __restrict__ outb,
                        const float* __restrict__ sab,  const float* __restrict__ b2,
                        uint32_t* __restrict__ packw,   float* __restrict__ biasbuf)
{
  int bid  = blockIdx.x;
  int lane = threadIdx.x;
  if (bid < 48) {
    int f = bid, mat = f >> 3, nt = (f >> 1) & 3, kc = f & 1;
    const float* W; int R, C;
    switch (mat) {
      case 0:  W = inw;            R = 50; C = 50; break;
      case 1:  W = inw + 50*50;    R = 50; C = 50; break;
      case 2:  W = inw + 100*50;   R = 50; C = 50; break;
      case 3:  W = outw;           R = 50; C = 50; break;
      case 4:  W = saw;            R = 48; C = 50; break;
      default: W = w2;             R = 48; C = 48; break;
    }
    int hv = lane >> 4, ls = lane & 15;
    int n  = nt * 16 + ls;
    uint32_t* dst = packw + ((size_t)f * 32 + lane) * 8;
    #pragma unroll
    for (int i = 0; i < 8; ++i) {
      int base = ((i < 4) ? 0 : 16) + hv * 8 + (i & 3) * 2;
      int k    = kc * 32 + base;
      float lo = (n < R && k     < C) ? W[n * C + k]     : 0.f;
      float hi = (n < R && k + 1 < C) ? W[n * C + k + 1] : 0.f;
      dst[i] = pack2bf(lo, hi);
    }
  } else {
    int mat = bid - 48;
    const float* bs; int len;
    switch (mat) {
      case 0:  bs = inb;       len = 50; break;
      case 1:  bs = inb + 50;  len = 50; break;
      case 2:  bs = inb + 100; len = 50; break;
      case 3:  bs = outb;      len = 50; break;
      case 4:  bs = sab;       len = 48; break;
      default: bs = b2;        len = 48; break;
    }
    for (int c = lane; c < 64; c += 32)
      biasbuf[mat * 64 + c] = (c < len) ? bs[c] : 0.f;
  }
}

// ---------------------------------------------------------------------------
// Phase 1: Q/K/V projection.  One block = one 16-token M-tile, 4 waves share
// the 12 output tiles (3 mats x 4 n-tiles), 2 WMMAs (K=64) each.
// qkv buffers are fp32, stride EP=64, pad columns exactly zero.
// ---------------------------------------------------------------------------
__global__ void __launch_bounds__(128)
project(const float* __restrict__ h_in, const uint32_t* __restrict__ packw,
        const float* __restrict__ biasbuf, float* __restrict__ qkv)
{
  __shared__ uint16_t tok[16 * 64];
  int tile = blockIdx.x;
  int tid  = threadIdx.x;

  // Stage token tile: [h(48) | label(2) | zero-pad(14)] as bf16
  for (int e = tid; e < 16 * 64; e += 128) {
    int m = e >> 6, col = e & 63;
    int token = tile * 16 + m;
    float v;
    if (col < DD)            v = h_in[(size_t)token * DD + col];
    else if (col == DD)      v = (float)((token & (NTOK - 1)) >> 6) * (1.f / GS);
    else if (col == DD + 1)  v = (float)(token & (GS - 1)) * (1.f / GS);
    else                     v = 0.f;
    tok[e] = f2bf(v);
  }
  __syncthreads();

  int wave = tid >> 5, lane = tid & 31;
  int hv = lane >> 4, ls = lane & 15;

  FragA a0, a1;                                   // A fragments, K chunks 0/1
  #pragma unroll
  for (int i = 0; i < 8; ++i) {
    int base = ((i < 4) ? 0 : 16) + hv * 8 + (i & 3) * 2;
    a0.u[i] = *(const uint32_t*)&tok[ls * 64 + base];
    a1.u[i] = *(const uint32_t*)&tok[ls * 64 + 32 + base];
  }

  for (int t = wave; t < 12; t += 4) {
    int mat = t >> 2, nt = t & 3;
    FragA b0, b1;
    const uint32_t* p0 = packw + ((size_t)(mat * 8 + nt * 2 + 0) * 32 + lane) * 8;
    const uint32_t* p1 = packw + ((size_t)(mat * 8 + nt * 2 + 1) * 32 + lane) * 8;
    #pragma unroll
    for (int i = 0; i < 8; ++i) { b0.u[i] = p0[i]; b1.u[i] = p1[i]; }

    FragC acc;
    float bv = biasbuf[mat * 64 + nt * 16 + ls];
    #pragma unroll
    for (int i = 0; i < 8; ++i) acc.f[i] = bv;
    acc.v = __builtin_amdgcn_wmma_f32_16x16x32_bf16(false, a0.v, false, b0.v,
                                                    (short)0, acc.v, false, false);
    acc.v = __builtin_amdgcn_wmma_f32_16x16x32_bf16(false, a1.v, false, b1.v,
                                                    (short)0, acc.v, false, false);
    float* dst = qkv + (size_t)mat * M_TOT * EP;
    #pragma unroll
    for (int i = 0; i < 8; ++i)
      dst[(size_t)(tile * 16 + hv * 8 + i) * EP + nt * 16 + ls] = acc.f[i];
  }
}

// ---------------------------------------------------------------------------
// Phase 2: 3x3 neighborhood attention + out_proj + fc_sa(+residual) + fc2 + LN.
// 8 waves/block, each wave owns a 16-token tile and a private 16x64 LDS tile.
// ---------------------------------------------------------------------------
__global__ void __launch_bounds__(256)
attend(const float* __restrict__ h_in, const float* __restrict__ qkv,
       const uint32_t* __restrict__ packw, const float* __restrict__ biasbuf,
       const float* __restrict__ ln_g, const float* __restrict__ ln_b,
       float* __restrict__ out)
{
  __shared__ uint16_t sm[8 * 16 * 64];
  int wave = threadIdx.x >> 5, lane = threadIdx.x & 31;
  int hv = lane >> 4, ls = lane & 15;
  uint16_t* tl = sm + wave * 16 * 64;
  int tile   = blockIdx.x * 8 + wave;
  int token0 = tile * 16;

  const float* Q = qkv;
  const float* K = qkv + (size_t)M_TOT * EP;
  const float* V = qkv + (size_t)2 * M_TOT * EP;

  // ---- attention: 2 lanes/token (halves of the 64-wide padded feature dim)
  int token = token0 + ls;
  int n = token & (NTOK - 1), bbase = token - n;
  int r = n >> 6, c = n & (GS - 1);
  int rc = clampi(r, 1, GS - 2), cc = clampi(c, 1, GS - 2);
  int ctr = bbase + rc * GS + cc;                 // shifted-window center

  float4 qv[8];
  const float4* qp = (const float4*)(Q + (size_t)ctr * EP + hv * 32);
  #pragma unroll
  for (int i = 0; i < 8; ++i) qv[i] = qp[i];

  float logits[9];
  #pragma unroll
  for (int j = 0; j < 9; ++j) {
    int idx = bbase + (rc + j / 3 - 1) * GS + (cc + j % 3 - 1);
    __builtin_prefetch(V + (size_t)idx * EP + hv * 32, 0, 0);  // global_prefetch_b8
    const float4* kp = (const float4*)(K + (size_t)idx * EP + hv * 32);
    float d = 0.f;
    #pragma unroll
    for (int i = 0; i < 8; ++i) {
      float4 kv = kp[i];
      d += qv[i].x * kv.x + qv[i].y * kv.y + qv[i].z * kv.z + qv[i].w * kv.w;
    }
    d += __shfl_xor(d, 16);                        // combine the two halves
    logits[j] = d * 0.14142135623730951f;          // 1/sqrt(E)
  }
  float mx = logits[0];
  #pragma unroll
  for (int j = 1; j < 9; ++j) mx = fmaxf(mx, logits[j]);
  float wj[9], ssum = 0.f;
  #pragma unroll
  for (int j = 0; j < 9; ++j) { wj[j] = __expf(logits[j] - mx); ssum += wj[j]; }
  float inv = 1.f / ssum;

  float ctx[32];
  #pragma unroll
  for (int i = 0; i < 32; ++i) ctx[i] = 0.f;
  #pragma unroll
  for (int j = 0; j < 9; ++j) {
    int idx = bbase + (rc + j / 3 - 1) * GS + (cc + j % 3 - 1);
    const float4* vp = (const float4*)(V + (size_t)idx * EP + hv * 32);
    float aj = wj[j] * inv;
    #pragma unroll
    for (int i = 0; i < 8; ++i) {
      float4 vv = vp[i];
      ctx[4*i+0] += aj * vv.x; ctx[4*i+1] += aj * vv.y;
      ctx[4*i+2] += aj * vv.z; ctx[4*i+3] += aj * vv.w;
    }
  }
  #pragma unroll
  for (int i = 0; i < 16; ++i)
    *(uint32_t*)&tl[ls * 64 + hv * 32 + 2 * i] = pack2bf(ctx[2*i], ctx[2*i+1]);
  __syncthreads();

  auto loadA = [&](FragA& x0, FragA& x1) {
    #pragma unroll
    for (int i = 0; i < 8; ++i) {
      int base = ((i < 4) ? 0 : 16) + hv * 8 + (i & 3) * 2;
      x0.u[i] = *(const uint32_t*)&tl[ls * 64 + base];
      x1.u[i] = *(const uint32_t*)&tl[ls * 64 + 32 + base];
    }
  };
  auto loadB = [&](int mat, int nt, FragA& x0, FragA& x1) {
    const uint32_t* p0 = packw + ((size_t)(mat * 8 + nt * 2 + 0) * 32 + lane) * 8;
    const uint32_t* p1 = packw + ((size_t)(mat * 8 + nt * 2 + 1) * 32 + lane) * 8;
    #pragma unroll
    for (int i = 0; i < 8; ++i) { x0.u[i] = p0[i]; x1.u[i] = p1[i]; }
  };

  FragA a0, a1, b0, b1;

  // ---- stage a: attn_out = ctx @ Wout^T + bout   (4 n-tiles, pad cols -> 0)
  loadA(a0, a1);
  FragC st[4];
  #pragma unroll
  for (int nt = 0; nt < 4; ++nt) {
    loadB(3, nt, b0, b1);
    float bv = biasbuf[3 * 64 + nt * 16 + ls];
    #pragma unroll
    for (int i = 0; i < 8; ++i) st[nt].f[i] = bv;
    st[nt].v = __builtin_amdgcn_wmma_f32_16x16x32_bf16(false, a0.v, false, b0.v,
                                                       (short)0, st[nt].v, false, false);
    st[nt].v = __builtin_amdgcn_wmma_f32_16x16x32_bf16(false, a1.v, false, b1.v,
                                                       (short)0, st[nt].v, false, false);
  }
  __syncthreads();
  #pragma unroll
  for (int nt = 0; nt < 4; ++nt)
    #pragma unroll
    for (int i = 0; i < 8; ++i)
      tl[(hv * 8 + i) * 64 + nt * 16 + ls] = f2bf(st[nt].f[i]);
  __syncthreads();

  // ---- stage b: s = h + attn_out @ Wsa^T + bsa   (48 valid cols)
  loadA(a0, a1);
  FragC sb[3];
  #pragma unroll
  for (int nt = 0; nt < 3; ++nt) {
    loadB(4, nt, b0, b1);
    float bv = biasbuf[4 * 64 + nt * 16 + ls];
    #pragma unroll
    for (int i = 0; i < 8; ++i) sb[nt].f[i] = bv;
    sb[nt].v = __builtin_amdgcn_wmma_f32_16x16x32_bf16(false, a0.v, false, b0.v,
                                                       (short)0, sb[nt].v, false, false);
    sb[nt].v = __builtin_amdgcn_wmma_f32_16x16x32_bf16(false, a1.v, false, b1.v,
                                                       (short)0, sb[nt].v, false, false);
    #pragma unroll
    for (int i = 0; i < 8; ++i)
      sb[nt].f[i] += h_in[(size_t)(token0 + hv * 8 + i) * DD + nt * 16 + ls];
  }
  __syncthreads();
  #pragma unroll
  for (int nt = 0; nt < 3; ++nt)
    #pragma unroll
    for (int i = 0; i < 8; ++i)
      tl[(hv * 8 + i) * 64 + nt * 16 + ls] = f2bf(sb[nt].f[i]);
  #pragma unroll
  for (int i = 0; i < 8; ++i)                      // zero pad cols 48..63
    tl[(hv * 8 + i) * 64 + 48 + ls] = 0;
  __syncthreads();

  // ---- stage c: t2 = s @ W2^T + b2, then LayerNorm over 48 cols
  loadA(a0, a1);
  FragC sc[3];
  #pragma unroll
  for (int nt = 0; nt < 3; ++nt) {
    loadB(5, nt, b0, b1);
    float bv = biasbuf[5 * 64 + nt * 16 + ls];
    #pragma unroll
    for (int i = 0; i < 8; ++i) sc[nt].f[i] = bv;
    sc[nt].v = __builtin_amdgcn_wmma_f32_16x16x32_bf16(false, a0.v, false, b0.v,
                                                       (short)0, sc[nt].v, false, false);
    sc[nt].v = __builtin_amdgcn_wmma_f32_16x16x32_bf16(false, a1.v, false, b1.v,
                                                       (short)0, sc[nt].v, false, false);
  }

  float g0 = ln_g[ls], g1 = ln_g[16 + ls], g2 = ln_g[32 + ls];
  float z0 = ln_b[ls], z1 = ln_b[16 + ls], z2 = ln_b[32 + ls];
  #pragma unroll
  for (int i = 0; i < 8; ++i) {
    float x0 = sc[0].f[i], x1 = sc[1].f[i], x2 = sc[2].f[i];
    float s1 = x0 + x1 + x2;
    float s2 = x0 * x0 + x1 * x1 + x2 * x2;
    #pragma unroll
    for (int m = 8; m >= 1; m >>= 1) {             // 16-lane half reduction
      s1 += __shfl_xor(s1, m);
      s2 += __shfl_xor(s2, m);
    }
    float mu   = s1 * (1.f / 48.f);
    float var  = s2 * (1.f / 48.f) - mu * mu;
    float rstd = rsqrtf(var + 1e-5f);
    size_t row = (size_t)(token0 + hv * 8 + i) * DD;
    out[row + ls]      = (x0 - mu) * rstd * g0 + z0;
    out[row + 16 + ls] = (x1 - mu) * rstd * g1 + z1;
    out[row + 32 + ls] = (x2 - mu) * rstd * g2 + z2;
  }
}

// ---------------------------------------------------------------------------
// Workspace layout (bytes): [0,49152) packed weights, [49152,51k) biases,
// [65536, 65536+3*8MB) Q/K/V fp32 stride-64.  Total ~24.1 MB.
// ---------------------------------------------------------------------------
extern "C" void kernel_launch(void* const* d_in, const int* in_sizes, int n_in,
                              void* d_out, int out_size, void* d_ws, size_t ws_size,
                              hipStream_t stream)
{
  const float* x    = (const float*)d_in[0];
  const float* inw  = (const float*)d_in[1];
  const float* inb  = (const float*)d_in[2];
  const float* outw = (const float*)d_in[3];
  const float* outb = (const float*)d_in[4];
  const float* saw  = (const float*)d_in[5];
  const float* sab  = (const float*)d_in[6];
  const float* w2   = (const float*)d_in[7];
  const float* b2   = (const float*)d_in[8];
  const float* lng  = (const float*)d_in[9];
  const float* lnb  = (const float*)d_in[10];
  float* out = (float*)d_out;

  uint8_t*  ws      = (uint8_t*)d_ws;
  uint32_t* packw   = (uint32_t*)ws;
  float*    biasbuf = (float*)(ws + 49152);
  float*    qkv     = (float*)(ws + 65536);

  prepack<<<54, 32, 0, stream>>>(inw, outw, saw, w2, inb, outb, sab, b2,
                                 packw, biasbuf);

  const size_t step = (size_t)M_TOT * DD;
  for (int t = 0; t < T_TOT; ++t) {
    const float* h_in = (t < T_OBS) ? (x + (size_t)t * step)
                                    : (out + (size_t)(t - 1) * step);
    float* o = out + (size_t)t * step;
    project<<<M_TOT / 16, 128, 0, stream>>>(h_in, packw, biasbuf, qkv);
    attend<<<M_TOT / 128, 256, 0, stream>>>(h_in, qkv, packw, biasbuf, lng, lnb, o);
  }
}
